// GCN_50457275793566
// MI455X (gfx1250) — matrix operand order repaired and checked by reading, not compiled, not run on hardware
//
#include <hip/hip_runtime.h>
#include <hip/hip_bf16.h>

// ---------------------------------------------------------------------------
// MI455X (gfx1250) implementation.
// Compute-bound (~120 GFLOP vs ~0.5GB traffic) -> everything through
// v_wmma_f32_16x16x32_bf16 (K=32 per instr vs K=4 for f32 WMMA).
// All GEMM operands staged/stored as bf16; f32 accumulation.
// ---------------------------------------------------------------------------

typedef __attribute__((ext_vector_type(16))) __bf16 v16bf;
typedef __attribute__((ext_vector_type(8)))  __bf16 v8bf;
typedef __attribute__((ext_vector_type(8)))  float  v8f;

union Frag { v16bf v; v8bf h[2]; };

static __device__ inline v8f wmma_bf16(const Frag& a, const Frag& b, v8f c) {
  // (neg_a, A, neg_b, B, c_mod, C, reuse_a, reuse_b)
  return __builtin_amdgcn_wmma_f32_16x16x32_bf16(false, a.v, false, b.v,
                                                 (short)0, c, false, false);
}

static __device__ inline v8f zero8() {
  v8f z;
#pragma unroll
  for (int i = 0; i < 8; ++i) z[i] = 0.0f;
  return z;
}

#define B_   16
#define C_   64
#define W_   256
#define H_   128
#define HO_  122
#define KW_  7
#define KT_  448   // C_ * KW_

// ---------------------------------------------------------------------------
// Kernel 1: K & Q convs fused. One workgroup per (b,w); per-(b,w) GEMM is
// [64 co x 448 k] x [448 k x 128 n(h', padded from 122)].
// 4 waves, each owns a 16-row co tile, accumulates both convs (shared B tile).
// LDS: A tiles [2][64][32] bf16, B tile [128 n][32 k] bf16 (N-major so B
// fragments are two b128 reads per lane).
// ---------------------------------------------------------------------------
__global__ __launch_bounds__(128) void conv_qk_kernel(
    const float* __restrict__ x,
    const float* __restrict__ wK, const float* __restrict__ bK,
    const float* __restrict__ wQ, const float* __restrict__ bQ,
    __bf16* __restrict__ Kc, __bf16* __restrict__ Qc)
{
  __shared__ __bf16 lsA[2][64][40];   // pitch 40 halves: 16B-aligned rows
  __shared__ __bf16 lsB[128][40];

  const int tid  = threadIdx.x;
  const int lane = tid & 31;
  const int wave = tid >> 5;
  const int lhi  = lane >> 4;
  const int llo  = lane & 15;
  const int b    = blockIdx.x >> 8;
  const int w    = blockIdx.x & 255;
  const int m0   = wave * 16;

  v8f accK[8], accQ[8];
#pragma unroll
  for (int i = 0; i < 8; ++i) { accK[i] = zero8(); accQ[i] = zero8(); }

  for (int kc = 0; kc < 14; ++kc) {
    const int k0 = kc * 32;
    // stage weights (both convs): tid -> (cv, co-row), 32 k each
    {
      const int cv = tid >> 6;
      const int m  = tid & 63;
      const float* wr = (cv ? wQ : wK) + m * KT_ + k0;
#pragma unroll
      for (int j = 0; j < 32; ++j) lsA[cv][m][j] = (__bf16)wr[j];
    }
    // stage im2col B tile transposed to [n][k]: k -> (ci, tap)
    {
      const int kk  = tid >> 2;           // 0..31
      const int g   = tid & 3;            // 4 groups of 32 n
      const int k   = k0 + kk;
      const int ci  = k / KW_;
      const int tap = k - ci * KW_;
      const float* src = x + (((size_t)b * C_ + ci) * W_ + w) * H_;
      const int n0 = g * 32;
#pragma unroll
      for (int j = 0; j < 32; ++j) {
        const int n = n0 + j;
        const float v = (n < HO_) ? src[n + tap] : 0.0f;  // n<122 => n+tap<128
        lsB[n][kk] = (__bf16)v;
      }
    }
    __syncthreads();

    Frag aK, aQ;
    const int arow = m0 + llo;            // A: lane = M + 16*(K>>3 & 1)
    aK.h[0] = *(const v8bf*)&lsA[0][arow][lhi * 8];
    aK.h[1] = *(const v8bf*)&lsA[0][arow][16 + lhi * 8];
    aQ.h[0] = *(const v8bf*)&lsA[1][arow][lhi * 8];
    aQ.h[1] = *(const v8bf*)&lsA[1][arow][16 + lhi * 8];
#pragma unroll
    for (int nt = 0; nt < 8; ++nt) {
      Frag bf;                            // B: lane = N + 16*(K>>3 & 1)
      const int brow = nt * 16 + llo;
      bf.h[0] = *(const v8bf*)&lsB[brow][lhi * 8];
      bf.h[1] = *(const v8bf*)&lsB[brow][16 + lhi * 8];
      accK[nt] = wmma_bf16(aK, bf, accK[nt]);
      accQ[nt] = wmma_bf16(aQ, bf, accQ[nt]);
    }
    __syncthreads();
  }

  // epilogue: +bias, cvt bf16, store contiguous [b][co][w][h'] (=> the
  // python reshape to [B,256,7808] is a free reinterpretation)
#pragma unroll
  for (int nt = 0; nt < 8; ++nt) {
    const int hp = nt * 16 + llo;
    if (hp < HO_) {
#pragma unroll
      for (int r = 0; r < 8; ++r) {
        const int co = m0 + r + 8 * lhi;  // C/D: VGPR r -> M = r + 8*(lane>>4)
        const size_t oidx = (((size_t)b * C_ + co) * W_ + w) * HO_ + hp;
        Kc[oidx] = (__bf16)(accK[nt][r] + bK[co]);
        Qc[oidx] = (__bf16)(accQ[nt][r] + bQ[co]);
      }
    }
  }
}

// ---------------------------------------------------------------------------
// Kernel 2: scores = Q K^T / sqrt(7808) (M=N=256, K=7808), block mask,
// softmax -> P (bf16). One workgroup (8 waves) per (b, 16-row i-tile); each
// wave owns two 16x16 j-tiles. Both fragments stream straight from global
// (per-lane chunks are contiguous 16B); per-batch Q+K = 8MB -> L2 resident.
// ---------------------------------------------------------------------------
__global__ __launch_bounds__(256) void attn_softmax_kernel(
    const __bf16* __restrict__ Qc, const __bf16* __restrict__ Kc,
    __bf16* __restrict__ Pb)
{
  __shared__ float sc[16][256];
  __shared__ float red[16][16];

  const int tid  = threadIdx.x;
  const int lane = tid & 31;
  const int wave = tid >> 5;
  const int lhi  = lane >> 4;
  const int llo  = lane & 15;
  const int b    = blockIdx.x >> 4;
  const int i0   = (blockIdx.x & 15) * 16;

  v8f acc[2];
  acc[0] = zero8(); acc[1] = zero8();

  const size_t qbase = ((size_t)b * 256 + i0 + llo) * 7808;
#pragma unroll 1
  for (int kc = 0; kc < 244; ++kc) {
    const int k0 = kc * 32;
    Frag a;
    a.h[0] = *(const v8bf*)(Qc + qbase + k0 + lhi * 16);
    a.h[1] = *(const v8bf*)(Qc + qbase + k0 + lhi * 16 + 8);
#pragma unroll
    for (int t = 0; t < 2; ++t) {
      const int j0 = (wave * 2 + t) * 16;
      const size_t kbase = ((size_t)b * 256 + j0 + llo) * 7808 + k0 + lhi * 16;
      Frag bf;
      bf.h[0] = *(const v8bf*)(Kc + kbase);
      bf.h[1] = *(const v8bf*)(Kc + kbase + 8);
      acc[t] = wmma_bf16(a, bf, acc[t]);
    }
  }

  const float scale = rsqrtf(7808.0f);
#pragma unroll
  for (int t = 0; t < 2; ++t) {
    const int j = (wave * 2 + t) * 16 + llo;
#pragma unroll
    for (int r = 0; r < 8; ++r) {
      const int row = r + 8 * lhi;
      const int i = i0 + row;
      const bool blk = ((i < 128) && (j < 128)) || ((i >= 128) && (j >= 128));
      sc[row][j] = blk ? -__builtin_inff() : acc[t][r] * scale;
    }
  }
  __syncthreads();

  // softmax: 16 threads per row
  const int row = tid >> 4;
  const int seg = tid & 15;
  float m = -__builtin_inff();
#pragma unroll
  for (int jj = 0; jj < 16; ++jj) m = fmaxf(m, sc[row][seg * 16 + jj]);
  red[row][seg] = m;
  __syncthreads();
  float rm = -__builtin_inff();
#pragma unroll
  for (int s = 0; s < 16; ++s) rm = fmaxf(rm, red[row][s]);
  float e[16];
  float lsum = 0.0f;
#pragma unroll
  for (int jj = 0; jj < 16; ++jj) {
    e[jj] = __expf(sc[row][seg * 16 + jj] - rm);  // exp(-inf)=0 handles mask
    lsum += e[jj];
  }
  __syncthreads();
  red[row][seg] = lsum;
  __syncthreads();
  float rs = 0.0f;
#pragma unroll
  for (int s = 0; s < 16; ++s) rs += red[row][s];
  const float inv = 1.0f / rs;
  const size_t pbase = ((size_t)b * 256 + i0 + row) * 256 + seg * 16;
#pragma unroll
  for (int jj = 0; jj < 16; ++jj) Pb[pbase + jj] = (__bf16)(e[jj] * inv);
}

// ---------------------------------------------------------------------------
// Kernel 3: x1 = P @ V (M=256, N=8192, K=256 per batch) with V = raw view of
// x as [256][8192]; then xs = bf16(x1 + x) stored [b][c][w][h] (again, the
// reshape back to NCHW is free). Workgroup: (b, i-tile, 256-col n-strip);
// 8 waves x 2 n-tiles. V k-chunks staged to LDS N-major with f32->bf16 cvt.
// ---------------------------------------------------------------------------
__global__ __launch_bounds__(256) void pv_add_kernel(
    const __bf16* __restrict__ Pb, const float* __restrict__ x,
    __bf16* __restrict__ xs)
{
  __shared__ __bf16 vt[256][40];

  const int tid  = threadIdx.x;
  const int lane = tid & 31;
  const int wave = tid >> 5;
  const int lhi  = lane >> 4;
  const int llo  = lane & 15;
  const int bid  = blockIdx.x;
  const int ng   = bid & 31;            // 32 strips of 256 columns
  const int it   = (bid >> 5) & 15;
  const int b    = bid >> 9;
  const int i0   = it * 16;
  const int n0   = ng * 256;

  v8f acc[2];
  acc[0] = zero8(); acc[1] = zero8();

  const size_t pbase = ((size_t)b * 256 + i0 + llo) * 256;
  const size_t xbat  = (size_t)b * 2097152;   // 64*256*128

  for (int kc = 0; kc < 8; ++kc) {
    const int k0 = kc * 32;
    {
      const int kk = tid >> 3;            // 0..31
      const int g  = tid & 7;             // 8 groups of 32 n
      const float* src = x + xbat + (size_t)(k0 + kk) * 8192 + n0 + g * 32;
#pragma unroll
      for (int j = 0; j < 32; ++j) vt[g * 32 + j][kk] = (__bf16)src[j];
    }
    __syncthreads();
    Frag a;
    a.h[0] = *(const v8bf*)(Pb + pbase + k0 + lhi * 16);
    a.h[1] = *(const v8bf*)(Pb + pbase + k0 + lhi * 16 + 8);
#pragma unroll
    for (int t = 0; t < 2; ++t) {
      const int brow = (wave * 2 + t) * 16 + llo;
      Frag bf;
      bf.h[0] = *(const v8bf*)&vt[brow][lhi * 16];
      bf.h[1] = *(const v8bf*)&vt[brow][lhi * 16 + 8];
      acc[t] = wmma_bf16(a, bf, acc[t]);
    }
    __syncthreads();
  }

#pragma unroll
  for (int t = 0; t < 2; ++t) {
    const int n = n0 + (wave * 2 + t) * 16 + llo;
#pragma unroll
    for (int r = 0; r < 8; ++r) {
      const int q = i0 + r + 8 * lhi;
      const size_t idx = xbat + (size_t)q * 8192 + n;
      xs[idx] = (__bf16)(acc[t][r] + x[idx]);   // residual add, cvt to bf16
    }
  }
}

// ---------------------------------------------------------------------------
// Kernel 4: final conv with wV on xs (already bf16) + bias + LeakyReLU(0.2),
// f32 output. Same structure as kernel 1, single conv.
// ---------------------------------------------------------------------------
__global__ __launch_bounds__(128) void conv_v_kernel(
    const __bf16* __restrict__ xs,
    const float* __restrict__ wV, const float* __restrict__ bV,
    float* __restrict__ out)
{
  __shared__ __bf16 lsA[64][40];
  __shared__ __bf16 lsB[128][40];

  const int tid  = threadIdx.x;
  const int lane = tid & 31;
  const int wave = tid >> 5;
  const int lhi  = lane >> 4;
  const int llo  = lane & 15;
  const int b    = blockIdx.x >> 8;
  const int w    = blockIdx.x & 255;
  const int m0   = wave * 16;

  v8f acc[8];
#pragma unroll
  for (int i = 0; i < 8; ++i) acc[i] = zero8();

  for (int kc = 0; kc < 14; ++kc) {
    const int k0 = kc * 32;
    {
      const int m = tid >> 1;
      const int g = tid & 1;
      const float* wr = wV + m * KT_ + k0 + g * 16;
#pragma unroll
      for (int j = 0; j < 16; ++j) lsA[m][g * 16 + j] = (__bf16)wr[j];
    }
    {
      const int kk  = tid >> 2;
      const int g   = tid & 3;
      const int k   = k0 + kk;
      const int ci  = k / KW_;
      const int tap = k - ci * KW_;
      const __bf16* src = xs + (((size_t)b * C_ + ci) * W_ + w) * H_;
      const int n0 = g * 32;
#pragma unroll
      for (int j = 0; j < 32; ++j) {
        const int n = n0 + j;
        lsB[n][kk] = (n < HO_) ? src[n + tap] : (__bf16)0.0f;
      }
    }
    __syncthreads();

    Frag a;
    const int arow = m0 + llo;
    a.h[0] = *(const v8bf*)&lsA[arow][lhi * 16];
    a.h[1] = *(const v8bf*)&lsA[arow][lhi * 16 + 8];
#pragma unroll
    for (int nt = 0; nt < 8; ++nt) {
      const int brow = nt * 16 + llo;
      Frag bf;
      bf.h[0] = *(const v8bf*)&lsB[brow][lhi * 16];
      bf.h[1] = *(const v8bf*)&lsB[brow][lhi * 16 + 8];
      acc[nt] = wmma_bf16(a, bf, acc[nt]);
    }
    __syncthreads();
  }

#pragma unroll
  for (int nt = 0; nt < 8; ++nt) {
    const int hp = nt * 16 + llo;
    if (hp < HO_) {
#pragma unroll
      for (int r = 0; r < 8; ++r) {
        const int co = m0 + r + 8 * lhi;
        float y = acc[nt][r] + bV[co];
        y = (y >= 0.0f) ? y : 0.2f * y;   // LeakyReLU(0.2)
        out[(((size_t)b * C_ + co) * W_ + w) * HO_ + hp] = y;
      }
    }
  }
}

// ---------------------------------------------------------------------------
// Host launcher. Workspace layout (bytes):
//   Kc  bf16 [16][64][256][122] :      0 .. 63,963,136
//   Qc  bf16 same               : 63,963,136 .. 127,926,272
//   P   bf16 [16][256][256]     : 127,926,272 .. 130,023,424
//   xs  bf16 [16][64][256][128] : 130,023,424 .. 197,132,288 (~188 MB)
// ---------------------------------------------------------------------------
extern "C" void kernel_launch(void* const* d_in, const int* in_sizes, int n_in,
                              void* d_out, int out_size, void* d_ws, size_t ws_size,
                              hipStream_t stream) {
  (void)in_sizes; (void)n_in; (void)out_size; (void)ws_size;
  const float* x  = (const float*)d_in[0];
  const float* wK = (const float*)d_in[1];
  const float* bK = (const float*)d_in[2];
  const float* wQ = (const float*)d_in[3];
  const float* bQ = (const float*)d_in[4];
  const float* wV = (const float*)d_in[5];
  const float* bV = (const float*)d_in[6];
  float* out = (float*)d_out;

  char* ws = (char*)d_ws;
  __bf16* Kc = (__bf16*)(ws);
  __bf16* Qc = (__bf16*)(ws + 63963136u);
  __bf16* Pb = (__bf16*)(ws + 127926272u);
  __bf16* xs = (__bf16*)(ws + 130023424u);

  conv_qk_kernel<<<dim3(B_ * W_), dim3(128), 0, stream>>>(x, wK, bK, wQ, bQ, Kc, Qc);
  attn_softmax_kernel<<<dim3(B_ * 16), dim3(256), 0, stream>>>(Qc, Kc, Pb);
  pv_add_kernel<<<dim3(B_ * 16 * 32), dim3(256), 0, stream>>>(Pb, x, xs);
  conv_v_kernel<<<dim3(B_ * W_), dim3(128), 0, stream>>>(xs, wV, bV, out);
}
